// GNN_65712999629491
// MI455X (gfx1250) — matrix operand order repaired and checked by reading, not compiled, not run on hardware
//
#include <hip/hip_runtime.h>
#include <hip/hip_bf16.h>

// MI455X / gfx1250. wave32. WMMA f16 -> f32 accumulate for the dense layers.
//
// Pipeline (all on `stream`):
//   1. deg=0; edge_prep: per-edge alpha scale + float degree count (atomic)
//   2. inv_deg in place
//   3. pack W1/W2/lin_w into per-lane WMMA B-fragment layout (f16), once
//   4. per layer: zero accum; wave-per-edge gather/scale/scatter (f32 atomics,
//      L2-resident: h=40MB, accum=40MB, both << 192MB L2);
//      then fused invdeg-scale + GEMM(v_wmma_f32_16x16x32_f16) + bias + relu
//   5. classifier GEMM (no invdeg, no relu) straight into d_out
//
// Workspace: E + N + 2*N*100 floats + packed weights  ~= 84 MB.

typedef __attribute__((ext_vector_type(16))) _Float16 v16h;
typedef __attribute__((ext_vector_type(8)))  float    v8f;

#define NNODES 100000
#define NEDGES 800000
#define FEATS  100
#define HID    100
#define CLS    50
#define GENE   20000
#define KPAD   128
#define KCH    4          // 128 / 32
#define ASTRIDE 132       // LDS row stride in halves (pad vs 128 to avoid bank conflicts)

__global__ void zero_f32(float* __restrict__ p, long n) {
  long i = (long)blockIdx.x * blockDim.x + threadIdx.x;
  long st = (long)gridDim.x * blockDim.x;
  for (; i < n; i += st) p[i] = 0.0f;
}

__global__ void edge_prep(const int* __restrict__ ids, const int* __restrict__ src,
                          const int* __restrict__ dst, const float* __restrict__ ew,
                          const float* __restrict__ alpha, float* __restrict__ scale,
                          float* __restrict__ deg) {
  int e = blockIdx.x * blockDim.x + threadIdx.x;
  if (e >= NEDGES) return;
  int sid = ids[src[e]];
  int did = ids[dst[e]];
  int idx = GENE + 1;
  if (sid >= 0 && did <  0) idx = sid;
  if (did >= 0 && sid <  0) idx = did;
  if (sid >= 0 && did >= 0) idx = GENE;
  scale[e] = alpha[idx] * ew[e];
  atomicAdd(&deg[dst[e]], 1.0f);   // integer-valued float sum: order-independent
}

__global__ void inv_deg_k(float* __restrict__ deg) {
  int n = blockIdx.x * blockDim.x + threadIdx.x;
  if (n >= NNODES) return;
  float d = deg[n];
  deg[n] = d > 0.0f ? 1.0f / d : 0.0f;
}

// Pack W[j,k] (row-major [outdim,indim]) into per-lane WMMA B fragments (f16).
// B layout (this file's contiguous K order, same as A fragment; KxN=32x16): column N = lane&15;
// lanes 0-15 hold K=0..15 (VGPR v -> K=2v,2v+1), lanes 16-31 hold K=16..31.
// Fragment index = (jt*KCH + c)*32 + lane, 16 contiguous halves each.
__global__ void pack_w(const float* __restrict__ W, _Float16* __restrict__ out,
                       int outdim, int indim, int jtiles) {
  int t = blockIdx.x * blockDim.x + threadIdx.x;
  if (t >= jtiles * KCH * 32) return;
  int lane = t & 31;
  int c    = (t >> 5) & 3;
  int jt   = t >> 7;
  int j    = jt * 16 + (lane & 15);
  int half = lane >> 4;
  _Float16* o = out + (size_t)t * 16;
#pragma unroll
  for (int v = 0; v < 8; ++v) {
#pragma unroll
    for (int p = 0; p < 2; ++p) {
      int k = c * 32 + half * 16 + v * 2 + p;
      float val = (j < outdim && k < indim) ? W[(size_t)j * indim + k] : 0.0f;
      o[v * 2 + p] = (_Float16)val;
    }
  }
}

// One wave (32 lanes) per edge; lane handles feats lane, lane+32, lane+64, lane+96.
// Gather from h (L2 resident) and atomic-add into accum (L2 RMW).
// Launched with exactly NEDGES*32 threads (multiple of 256) -> no stride loop.
__global__ void edge_scatter(const float* __restrict__ h, const int* __restrict__ src,
                             const int* __restrict__ dst, const float* __restrict__ scale,
                             float* __restrict__ acc) {
  long g   = (long)blockIdx.x * blockDim.x + threadIdx.x;
  int e    = (int)(g >> 5);
  int lane = (int)(g & 31);
  int s = src[e], d = dst[e];
  float sc = scale[e];
  const float* hs = h   + (size_t)s * FEATS;
  float*       ad = acc + (size_t)d * FEATS;
#pragma unroll
  for (int f = lane; f < FEATS; f += 32)
    atomicAdd(&ad[f], hs[f] * sc);
}

// out[n,j] = act( (USE_INVDEG ? invdeg[n] : 1) * sum_k A[n,k] * W[j,k] + bias[j] )
// Block = one 16-node M-tile (NNODES % 16 == 0 -> no M bounds checks anywhere);
// wave w computes the 16-wide output tile jt=w. K padded to 128, 4 WMMA chunks.
template <bool USE_INVDEG, bool RELU>
__global__ void gemm_wmma(const float* __restrict__ A, const float* __restrict__ invdeg,
                          const _Float16* __restrict__ Bpack, const float* __restrict__ bias,
                          float* __restrict__ out, int outdim, int jtiles) {
  __shared__ _Float16 A_sh[16 * ASTRIDE];
  int base = blockIdx.x * 16;
  int tid  = threadIdx.x;

  // Stage A tile: f32 -> f16, fused inv-degree scaling, zero-pad K to 128.
  for (int idx = tid; idx < 16 * KPAD; idx += blockDim.x) {
    int r = idx >> 7, c = idx & (KPAD - 1);
    int node = base + r;
    float v = 0.0f;
    if (c < FEATS) {
      v = A[(size_t)node * FEATS + c];
      if (USE_INVDEG) v *= invdeg[node];
    }
    A_sh[r * ASTRIDE + c] = (_Float16)v;
  }
  __syncthreads();

  int wave = tid >> 5, lane = tid & 31;
  if (wave >= jtiles) return;          // wave-uniform exit: EXEC all-1s for WMMA
  int jt   = wave;
  int m    = lane & 15;                // A row for this lane; also C/D column
  int half = lane >> 4;
  int j    = jt * 16 + m;

  v8f acc = {};
#pragma unroll
  for (int c = 0; c < KCH; ++c) {
    // A fragment (contiguous K order, matching B pack): lanes 0-15 K=0-15; lanes 16-31 K=16-31
    v16h a;
    const _Float16* as = &A_sh[m * ASTRIDE + c * 32];
#pragma unroll
    for (int v = 0; v < 8; ++v) {
      int klb = v * 2 + half * 16;
      a[2 * v]     = as[klb];
      a[2 * v + 1] = as[klb + 1];
    }
    // B fragment: contiguous 32B per lane, pre-packed
    v16h b = *(const v16h*)(Bpack + ((size_t)(jt * KCH + c) * 32 + lane) * 16);
    acc = __builtin_amdgcn_wmma_f32_16x16x32_f16(false, a, false, b,
                                                 (short)0, acc, false, false);
  }

  // Epilogue: single per-lane column guard, then 8 straight-line stores.
  // C/D layout: VGPR r -> row (r + 8*half), col = lane&15.
  if (j < outdim) {
    float bv = bias[j];
    float* op = out + (size_t)(base + half * 8) * outdim + j;
#pragma unroll
    for (int r = 0; r < 8; ++r) {
      float v = acc[r] + bv;
      if (RELU) v = fmaxf(v, 0.0f);
      op[(size_t)r * outdim] = v;
    }
  }
}

extern "C" void kernel_launch(void* const* d_in, const int* in_sizes, int n_in,
                              void* d_out, int out_size, void* d_ws, size_t ws_size,
                              hipStream_t stream) {
  const float* features = (const float*)d_in[0];
  const int*   node_ids = (const int*)  d_in[1];
  const int*   src      = (const int*)  d_in[2];
  const int*   dst      = (const int*)  d_in[3];
  const float* ew       = (const float*)d_in[4];
  const float* alpha    = (const float*)d_in[5];
  const float* W1       = (const float*)d_in[6];
  const float* b1       = (const float*)d_in[7];
  const float* W2       = (const float*)d_in[8];
  const float* b2       = (const float*)d_in[9];
  const float* lw       = (const float*)d_in[10];
  const float* lb       = (const float*)d_in[11];
  float* outp = (float*)d_out;

  // Workspace carve-up (prefix byte counts keep wp* 32B aligned).
  float* ws    = (float*)d_ws;
  float* scale = ws;                                    // E
  float* deg   = scale + NEDGES;                        // N  (becomes inv_deg)
  float* buf0  = deg + NNODES;                          // N*HID
  float* buf1  = buf0 + (size_t)NNODES * HID;           // N*HID
  _Float16* wp1 = (_Float16*)(buf1 + (size_t)NNODES * HID);
  _Float16* wp2 = wp1 + 7 * KCH * 32 * 16;
  _Float16* wpl = wp2 + 7 * KCH * 32 * 16;

  const int T = 256;
  zero_f32 <<<1024, T, 0, stream>>>(deg, NNODES);
  edge_prep<<<(NEDGES + T - 1) / T, T, 0, stream>>>(node_ids, src, dst, ew, alpha, scale, deg);
  inv_deg_k<<<(NNODES + T - 1) / T, T, 0, stream>>>(deg);
  pack_w   <<<(7 * KCH * 32 + T - 1) / T, T, 0, stream>>>(W1, wp1, HID, FEATS, 7);
  pack_w   <<<(7 * KCH * 32 + T - 1) / T, T, 0, stream>>>(W2, wp2, HID, HID, 7);
  pack_w   <<<(4 * KCH * 32 + T - 1) / T, T, 0, stream>>>(lw, wpl, CLS, HID, 4);

  const long nb = (long)NNODES * HID;
  const int scatter_blocks = (int)(((long)NEDGES * 32) / T);   // exact multiple

  // Layer 1
  zero_f32    <<<4096, T, 0, stream>>>(buf1, nb);
  edge_scatter<<<scatter_blocks, T, 0, stream>>>(features, src, dst, scale, buf1);
  gemm_wmma<true, true><<<NNODES / 16, 224, 0, stream>>>(buf1, deg, wp1, b1, buf0, HID, 7);
  // Layer 2
  zero_f32    <<<4096, T, 0, stream>>>(buf1, nb);
  edge_scatter<<<scatter_blocks, T, 0, stream>>>(buf0, src, dst, scale, buf1);
  gemm_wmma<true, true><<<NNODES / 16, 224, 0, stream>>>(buf1, deg, wp2, b2, buf0, HID, 7);
  // Classifier: no inv-deg, no relu, straight to d_out
  gemm_wmma<false, false><<<NNODES / 16, 128, 0, stream>>>(buf0, nullptr, wpl, lb, outp, CLS, 4);
}
